// TransformerEncoder_31653908971926
// MI455X (gfx1250) — matrix operand order repaired and checked by reading, not compiled, not run on hardware
//
#include <hip/hip_runtime.h>
#include <hip/hip_bf16.h>
#include <math.h>

// ---------------- problem constants (from reference) ----------------
constexpr int Bc = 4, Sc = 2048, Dc = 1024, Hc = 16, FFc = 4096;
constexpr int DHc = Dc / Hc;          // 64
constexpr int Mrows = Bc * Sc;        // 8192
constexpr float EPSc = 1e-6f;

// ---------------- WMMA types ----------------
typedef __attribute__((ext_vector_type(16))) __bf16 v16bf;
typedef __attribute__((ext_vector_type(8)))  float  v8f;

// ---------------- gfx1250 async global->LDS staging (guarded) ----------------
#if defined(__has_builtin)
#if __has_builtin(__builtin_amdgcn_global_load_async_to_lds_b128)
#define USE_ASYNC_LDS 1
#endif
#endif
#ifndef USE_ASYNC_LDS
#define USE_ASYNC_LDS 0
#endif

// builtin prototype is (v4i addrspace(1)*, v4i addrspace(3)*, imm int, imm int)
// with v4i = GCC-style vector of 4 ints (per hipcc diagnostic).
typedef int v4i __attribute__((vector_size(16)));
typedef __attribute__((address_space(1))) v4i g_v4i;
typedef __attribute__((address_space(3))) v4i l_v4i;

// copy 16 bytes (8 bf16) global -> LDS, per lane
__device__ __forceinline__ void copy16_to_lds(unsigned short* ldst, const unsigned short* gsrc) {
#if USE_ASYNC_LDS
  __builtin_amdgcn_global_load_async_to_lds_b128((g_v4i*)gsrc, (l_v4i*)ldst, 0, 0);
#else
  *(uint4*)ldst = *(const uint4*)gsrc;
#endif
}
__device__ __forceinline__ void wait_lds_copies() {
#if USE_ASYNC_LDS
#if __has_builtin(__builtin_amdgcn_s_wait_asynccnt)
  __builtin_amdgcn_s_wait_asynccnt(0);
#else
  asm volatile("s_wait_asynccnt 0x0" ::: "memory");
#endif
#endif
}

__device__ __forceinline__ unsigned short f2bf(float f) {
  unsigned int u = __float_as_uint(f);
  u += 0x7FFFu + ((u >> 16) & 1u);            // round-to-nearest-even
  return (unsigned short)(u >> 16);
}
__device__ __forceinline__ __bf16 us2bf(unsigned short s) {
  return __builtin_bit_cast(__bf16, s);
}

__device__ __forceinline__ v8f wmma_bf16(v16bf a, v16bf b, v8f c) {
  return __builtin_amdgcn_wmma_f32_16x16x32_bf16(
      /*neg_a=*/false, a, /*neg_b=*/false, b,
      /*c_mod=*/(short)0, c, /*reuse_a=*/false, /*reuse_b=*/false);
}

// ---- fragment builders (wave32), contiguous k = 16*(lane/16) + s in both A and B, not the ISA table order ----
// A (16x32 MxK): lane 0-15 -> M=lane, K=0..15; lanes 16-31 -> same M, K=16..31.
// VGPR v: K=(half16)+2v+pos
__device__ __forceinline__ v16bf load_frag_a(const unsigned short* base, int lane, int ld) {
  const int m = lane & 15;
  const int khalf = (lane >> 4) * 16;
  const unsigned short* row = base + m * ld;
  v16bf a;
#pragma unroll
  for (int v = 0; v < 8; ++v) {
    int kbase = khalf + 2 * v;
    unsigned int pair = *(const unsigned int*)(row + kbase);
    a[2 * v]     = us2bf((unsigned short)(pair & 0xffffu));
    a[2 * v + 1] = us2bf((unsigned short)(pair >> 16));
  }
  return a;
}
// B (32x16 KxN) from row-major [K][N] tile: lane 0-15 -> N=lane, K=0..15; lanes 16-31 -> K=16..31
__device__ __forceinline__ v16bf load_frag_b(const unsigned short* base, int lane, int ld) {
  const int n = lane & 15;
  const int koff = (lane >> 4) * 16;
  v16bf b;
#pragma unroll
  for (int e = 0; e < 16; ++e)
    b[e] = us2bf(base[(size_t)(koff + e) * ld + n]);
  return b;
}
// B = T^t from a row-major [N][K] tile (used for Q@K^T): contiguous in K -> pair loads
__device__ __forceinline__ v16bf load_frag_bT(const unsigned short* base, int lane, int ld) {
  const int n = lane & 15;
  const int koff = (lane >> 4) * 16;
  const unsigned short* row = base + n * ld + koff;
  v16bf b;
#pragma unroll
  for (int v = 0; v < 8; ++v) {
    unsigned int pair = *(const unsigned int*)(row + 2 * v);
    b[2 * v]     = us2bf((unsigned short)(pair & 0xffffu));
    b[2 * v + 1] = us2bf((unsigned short)(pair >> 16));
  }
  return b;
}

// -------- wave32 half-wave (16-lane) reductions; xor masks <16 stay in half --------
__device__ __forceinline__ float half16_max(float v) {
#pragma unroll
  for (int m = 8; m >= 1; m >>= 1) v = fmaxf(v, __shfl_xor(v, m, 32));
  return v;
}
__device__ __forceinline__ float half16_sum(float v) {
#pragma unroll
  for (int m = 8; m >= 1; m >>= 1) v += __shfl_xor(v, m, 32);
  return v;
}

// ======================= fp32 -> bf16 conversion =======================
__global__ __launch_bounds__(256) void cvt_f32_bf16(const float* __restrict__ in,
                                                    unsigned short* __restrict__ out, int n) {
  int i = (blockIdx.x * 256 + threadIdx.x) * 4;
  if (i + 3 < n) {
    float4 f = *(const float4*)(in + i);
    out[i + 0] = f2bf(f.x); out[i + 1] = f2bf(f.y);
    out[i + 2] = f2bf(f.z); out[i + 3] = f2bf(f.w);
  } else {
    for (; i < n; ++i) out[i] = f2bf(in[i]);
  }
}

// ======================= bf16 WMMA GEMM ================================
// C[M,N] = A[M,K] @ B[K,N] + bias, templated epilogue.
// Block tile 128x128, BK=32, 256 threads = 8 waves (2x4), each wave 64x32 (4x2 subtiles).
enum { EPI_F32 = 0, EPI_BF16 = 1, EPI_GELU_BF16 = 2 };

template <int EPI>
__global__ __launch_bounds__(256, 2) void gemm_bf16_wmma(
    const unsigned short* __restrict__ A, const unsigned short* __restrict__ Bm,
    const float* __restrict__ bias, float* __restrict__ Cf,
    unsigned short* __restrict__ Cb, int Mdim, int Ndim, int Kdim) {
  __shared__ unsigned short As[128][40];   // pad 32->40: 80B rows, bank-skewed, 16B aligned
  __shared__ unsigned short Bs[32][136];   // pad 128->136: 272B rows

  const int tid  = threadIdx.x;
  const int lane = tid & 31;
  const int wid  = tid >> 5;
  const int wm   = wid >> 2;   // 0..1
  const int wn   = wid & 3;    // 0..3
  const int bm   = blockIdx.y * 128;
  const int bn   = blockIdx.x * 128;

  v8f acc[4][2];
#pragma unroll
  for (int i = 0; i < 4; ++i)
#pragma unroll
    for (int j = 0; j < 2; ++j) acc[i][j] = (v8f){};

  for (int k0 = 0; k0 < Kdim; k0 += 32) {
    if (k0 + 32 < Kdim) {   // speculative prefetch of next k-tile -> global_prefetch_b8
      __builtin_prefetch(A + (size_t)(bm + (tid >> 1)) * Kdim + k0 + 32, 0, 1);
      __builtin_prefetch(Bm + (size_t)(k0 + 32 + (tid >> 4)) * Ndim + bn + (tid & 15) * 8, 0, 1);
    }
#pragma unroll
    for (int it = 0; it < 2; ++it) {        // A tile: 128x32 bf16, 16B per thread
      int idx = (it * 256 + tid) * 8;
      int r = idx >> 5, c = idx & 31;
      copy16_to_lds(&As[r][c], A + (size_t)(bm + r) * Kdim + k0 + c);
    }
#pragma unroll
    for (int it = 0; it < 2; ++it) {        // B tile: 32x128 bf16
      int idx = (it * 256 + tid) * 8;
      int r = idx >> 7, c = idx & 127;
      copy16_to_lds(&Bs[r][c], Bm + (size_t)(k0 + r) * Ndim + bn + c);
    }
    wait_lds_copies();    // drain ASYNCcnt before making tiles visible
    __syncthreads();

    v16bf af[4], bf_[2];
#pragma unroll
    for (int mi = 0; mi < 4; ++mi)
      af[mi] = load_frag_a(&As[wm * 64 + mi * 16][0], lane, 40);
#pragma unroll
    for (int ni = 0; ni < 2; ++ni)
      bf_[ni] = load_frag_b(&Bs[0][wn * 32 + ni * 16], lane, 136);
#pragma unroll
    for (int mi = 0; mi < 4; ++mi)
#pragma unroll
      for (int ni = 0; ni < 2; ++ni)
        acc[mi][ni] = wmma_bf16(af[mi], bf_[ni], acc[mi][ni]);
    __syncthreads();
  }

  // epilogue: C layout -> VGPR r holds row r (lanes 0-15) / r+8 (lanes 16-31), col = lane&15
  const int rbase = bm + wm * 64 + ((lane >> 4) * 8);
#pragma unroll
  for (int mi = 0; mi < 4; ++mi) {
#pragma unroll
    for (int ni = 0; ni < 2; ++ni) {
      int col = bn + wn * 32 + ni * 16 + (lane & 15);
      float bv = bias[col];
#pragma unroll
      for (int r = 0; r < 8; ++r) {
        int row = rbase + mi * 16 + r;
        float val = acc[mi][ni][r] + bv;
        if (EPI == EPI_GELU_BF16)
          val = 0.5f * val * (1.0f + erff(val * 0.70710678118654752f));
        if (EPI == EPI_F32) Cf[(size_t)row * Ndim + col] = val;
        else                Cb[(size_t)row * Ndim + col] = f2bf(val);
      }
    }
  }
}

// ======================= flash attention (bf16 WMMA) ====================
// grid: (S/64, H, B); 128 threads = 4 waves; each wave owns 16 query rows.
__global__ __launch_bounds__(128, 2) void flash_attn(
    const unsigned short* __restrict__ Q, const unsigned short* __restrict__ K,
    const unsigned short* __restrict__ V, const int* __restrict__ mask,
    unsigned short* __restrict__ O) {
  __shared__ unsigned short Qs[64][72];
  __shared__ unsigned short Ks[64][72];
  __shared__ unsigned short Vs[64][72];
  __shared__ unsigned short Ps[4][16][72];

  const int qt = blockIdx.x, h = blockIdx.y, b = blockIdx.z;
  const int q0 = qt * 64;
  const int tid = threadIdx.x, lane = tid & 31, wid = tid >> 5;

  // load 64x64 Q tile (heads are column slices of [B,S,D])
#pragma unroll
  for (int it = 0; it < 4; ++it) {
    int idx = (it * 128 + tid) * 8;
    int r = idx >> 6, c = idx & 63;
    copy16_to_lds(&Qs[r][c], Q + ((size_t)b * Sc + q0 + r) * Dc + h * DHc + c);
  }

  float mrow[8], lrow[8];
  v8f acco[4];
#pragma unroll
  for (int r = 0; r < 8; ++r) { mrow[r] = -3.0e38f; lrow[r] = 0.f; }
#pragma unroll
  for (int n = 0; n < 4; ++n) acco[n] = (v8f){};

  const unsigned short* Qrow = &Qs[wid * 16][0];
  const int half8 = (lane >> 4) * 8;

  for (int j = 0; j < Sc; j += 64) {
    __syncthreads();
#pragma unroll
    for (int it = 0; it < 4; ++it) {  // K block
      int idx = (it * 128 + tid) * 8;
      int r = idx >> 6, c = idx & 63;
      copy16_to_lds(&Ks[r][c], K + ((size_t)b * Sc + j + r) * Dc + h * DHc + c);
    }
#pragma unroll
    for (int it = 0; it < 4; ++it) {  // V block
      int idx = (it * 128 + tid) * 8;
      int r = idx >> 6, c = idx & 63;
      copy16_to_lds(&Vs[r][c], V + ((size_t)b * Sc + j + r) * Dc + h * DHc + c);
    }
    wait_lds_copies();    // covers Q tile on first iteration too (ASYNCcnt -> 0)
    __syncthreads();

    // S = Q(16x64) @ K^T(64x64): 2 k-steps x 4 n-subtiles
    v8f accs[4];
#pragma unroll
    for (int n = 0; n < 4; ++n) accs[n] = (v8f){};
#pragma unroll
    for (int kk = 0; kk < 2; ++kk) {
      v16bf a = load_frag_a(Qrow + kk * 32, lane, 72);
#pragma unroll
      for (int n = 0; n < 4; ++n) {
        v16bf bb = load_frag_bT(&Ks[n * 16][kk * 32], lane, 72);
        accs[n] = wmma_bf16(a, bb, accs[n]);
      }
    }

    // scale + mask + row max
    float rmax[8];
#pragma unroll
    for (int r = 0; r < 8; ++r) rmax[r] = -3.0e38f;
#pragma unroll
    for (int n = 0; n < 4; ++n) {
      int kcol = j + n * 16 + (lane & 15);
#pragma unroll
      for (int r = 0; r < 8; ++r) {
        int qrow = q0 + wid * 16 + half8 + r;
        int mv = mask[(size_t)b * Sc * Sc + (size_t)qrow * Sc + kcol];
        float s = (mv == 0) ? -1.0e9f : accs[n][r] * 0.125f;  // 1/sqrt(64)
        accs[n][r] = s;
        rmax[r] = fmaxf(rmax[r], s);
      }
    }
#pragma unroll
    for (int r = 0; r < 8; ++r) rmax[r] = half16_max(rmax[r]);

    float cf[8];
#pragma unroll
    for (int r = 0; r < 8; ++r) {
      float mn = fmaxf(mrow[r], rmax[r]);
      cf[r] = __expf(mrow[r] - mn);
      mrow[r] = mn;
    }

    float rsum[8];
#pragma unroll
    for (int r = 0; r < 8; ++r) rsum[r] = 0.f;
#pragma unroll
    for (int n = 0; n < 4; ++n) {
#pragma unroll
      for (int r = 0; r < 8; ++r) {
        float p = __expf(accs[n][r] - mrow[r]);
        rsum[r] += p;
        Ps[wid][half8 + r][n * 16 + (lane & 15)] = f2bf(p);   // per-wave, same-wave reuse
      }
    }
#pragma unroll
    for (int r = 0; r < 8; ++r) {
      rsum[r] = half16_sum(rsum[r]);
      lrow[r] = lrow[r] * cf[r] + rsum[r];
    }
#pragma unroll
    for (int n = 0; n < 4; ++n)
#pragma unroll
      for (int r = 0; r < 8; ++r) acco[n][r] *= cf[r];

    // O += P(16x64) @ V(64x64)
#pragma unroll
    for (int kk = 0; kk < 2; ++kk) {
      v16bf pa = load_frag_a(&Ps[wid][0][kk * 32], lane, 72);
#pragma unroll
      for (int n = 0; n < 4; ++n) {
        v16bf vb = load_frag_b(&Vs[kk * 32][n * 16], lane, 72);
        acco[n] = wmma_bf16(pa, vb, acco[n]);
      }
    }
  }

  // write normalized output (bf16) back to [B,S,D] with heads as column slices
#pragma unroll
  for (int n = 0; n < 4; ++n) {
    int dh = n * 16 + (lane & 15);
#pragma unroll
    for (int r = 0; r < 8; ++r) {
      int qrow = q0 + wid * 16 + half8 + r;
      float val = acco[n][r] / lrow[r];
      O[((size_t)b * Sc + qrow) * Dc + h * DHc + dh] = f2bf(val);
    }
  }
}

// ======================= residual + LayerNorm ==========================
__device__ __forceinline__ float block_sum256(float v, float* sh) {
#pragma unroll
  for (int m = 16; m >= 1; m >>= 1) v += __shfl_xor(v, m, 32);
  int lane = threadIdx.x & 31, wid = threadIdx.x >> 5;
  if (lane == 0) sh[wid] = v;
  __syncthreads();
  if (threadIdx.x < 32) {
    float t = (threadIdx.x < 8) ? sh[threadIdx.x] : 0.f;
#pragma unroll
    for (int m = 4; m >= 1; m >>= 1) t += __shfl_xor(t, m, 32);
    if (threadIdx.x == 0) sh[0] = t;
  }
  __syncthreads();
  float r = sh[0];
  __syncthreads();
  return r;
}

template <bool WRITE_BF16>
__global__ __launch_bounds__(256) void ln_residual(
    const float* __restrict__ res, const float* __restrict__ t,
    const float* __restrict__ g, const float* __restrict__ bb,
    float* __restrict__ outf, unsigned short* __restrict__ outb) {
  __shared__ float sh[8];
  const int row = blockIdx.x;
  const int tid = threadIdx.x;
  const float* tr = t + (size_t)row * Dc;

  float v[4]; float s = 0.f;
#pragma unroll
  for (int i = 0; i < 4; ++i) { v[i] = tr[tid + i * 256]; s += v[i]; }
  float mean = block_sum256(s, sh) * (1.0f / Dc);
  float q = 0.f;
#pragma unroll
  for (int i = 0; i < 4; ++i) { float d = v[i] - mean; q += d * d; }
  float var = block_sum256(q, sh) * (1.0f / Dc);
  float inv = rsqrtf(var + EPSc);

#pragma unroll
  for (int i = 0; i < 4; ++i) {
    int idx = tid + i * 256;
    float y = res[(size_t)row * Dc + idx] + (v[i] - mean) * inv * g[idx] + bb[idx];
    outf[(size_t)row * Dc + idx] = y;
    if (WRITE_BF16) outb[(size_t)row * Dc + idx] = f2bf(y);
  }
}

// ======================= host-side orchestration =======================
extern "C" void kernel_launch(void* const* d_in, const int* in_sizes, int n_in,
                              void* d_out, int out_size, void* d_ws, size_t ws_size,
                              hipStream_t stream) {
  const float* x    = (const float*)d_in[0];
  const int*   mask = (const int*)  d_in[1];
  const float* Wq   = (const float*)d_in[2];
  const float* bq   = (const float*)d_in[3];
  const float* Wk   = (const float*)d_in[4];
  const float* bk   = (const float*)d_in[5];
  const float* Wv   = (const float*)d_in[6];
  const float* bv   = (const float*)d_in[7];
  const float* Wo   = (const float*)d_in[8];
  const float* bo   = (const float*)d_in[9];
  const float* ln1g = (const float*)d_in[10];
  const float* ln1b = (const float*)d_in[11];
  const float* W1   = (const float*)d_in[12];
  const float* b1   = (const float*)d_in[13];
  const float* W2   = (const float*)d_in[14];
  const float* b2   = (const float*)d_in[15];
  const float* ln2g = (const float*)d_in[16];
  const float* ln2b = (const float*)d_in[17];
  float* out = (float*)d_out;

  char* ws = (char*)d_ws;
  size_t off = 0;
  auto alloc = [&](size_t bytes) -> char* {
    char* p = ws + off;
    off = (off + bytes + 255) & ~(size_t)255;
    return p;
  };
  unsigned short* xb    = (unsigned short*)alloc((size_t)Mrows * Dc * 2);
  unsigned short* wqb   = (unsigned short*)alloc((size_t)Dc * Dc * 2);
  unsigned short* wkb   = (unsigned short*)alloc((size_t)Dc * Dc * 2);
  unsigned short* wvb   = (unsigned short*)alloc((size_t)Dc * Dc * 2);
  unsigned short* wob   = (unsigned short*)alloc((size_t)Dc * Dc * 2);
  unsigned short* w1b   = (unsigned short*)alloc((size_t)Dc * FFc * 2);
  unsigned short* w2b   = (unsigned short*)alloc((size_t)FFc * Dc * 2);
  unsigned short* qb    = (unsigned short*)alloc((size_t)Mrows * Dc * 2);
  unsigned short* kb    = (unsigned short*)alloc((size_t)Mrows * Dc * 2);
  unsigned short* vb    = (unsigned short*)alloc((size_t)Mrows * Dc * 2);
  unsigned short* attnb = (unsigned short*)alloc((size_t)Mrows * Dc * 2);
  float*          tmpf  = (float*)alloc((size_t)Mrows * Dc * 4);
  float*          yf    = (float*)alloc((size_t)Mrows * Dc * 4);
  unsigned short* yb    = (unsigned short*)alloc((size_t)Mrows * Dc * 2);
  unsigned short* ff1b  = (unsigned short*)alloc((size_t)Mrows * FFc * 2);
  (void)ws_size; (void)n_in; (void)in_sizes; (void)out_size;

  // 1) fp32 -> bf16 conversions
  struct Cv { const float* s; unsigned short* d; int n; } cv[] = {
      {x, xb, Mrows * Dc}, {Wq, wqb, Dc * Dc}, {Wk, wkb, Dc * Dc},
      {Wv, wvb, Dc * Dc},  {Wo, wob, Dc * Dc}, {W1, w1b, Dc * FFc},
      {W2, w2b, FFc * Dc}};
  for (auto& c : cv)
    cvt_f32_bf16<<<(c.n + 1023) / 1024, 256, 0, stream>>>(c.s, c.d, c.n);

  // 2) QKV projections
  dim3 gQKV(Dc / 128, Mrows / 128);
  gemm_bf16_wmma<EPI_BF16><<<gQKV, 256, 0, stream>>>(xb, wqb, bq, nullptr, qb, Mrows, Dc, Dc);
  gemm_bf16_wmma<EPI_BF16><<<gQKV, 256, 0, stream>>>(xb, wkb, bk, nullptr, kb, Mrows, Dc, Dc);
  gemm_bf16_wmma<EPI_BF16><<<gQKV, 256, 0, stream>>>(xb, wvb, bv, nullptr, vb, Mrows, Dc, Dc);

  // 3) flash attention
  flash_attn<<<dim3(Sc / 64, Hc, Bc), 128, 0, stream>>>(qb, kb, vb, mask, attnb);

  // 4) output projection (f32 out)
  gemm_bf16_wmma<EPI_F32><<<gQKV, 256, 0, stream>>>(attnb, wob, bo, tmpf, nullptr, Mrows, Dc, Dc);

  // 5) sublayer 1: y = x + LN(attn_out)
  ln_residual<true><<<Mrows, 256, 0, stream>>>(x, tmpf, ln1g, ln1b, yf, yb);

  // 6) FFN up + exact GELU (bf16 out)
  gemm_bf16_wmma<EPI_GELU_BF16><<<dim3(FFc / 128, Mrows / 128), 256, 0, stream>>>(
      yb, w1b, b1, nullptr, ff1b, Mrows, FFc, Dc);

  // 7) FFN down (f32 out)
  gemm_bf16_wmma<EPI_F32><<<gQKV, 256, 0, stream>>>(ff1b, w2b, b2, tmpf, nullptr, Mrows, Dc, FFc);

  // 8) sublayer 2: out = y + LN(ff)
  ln_residual<false><<<Mrows, 256, 0, stream>>>(yf, tmpf, ln2g, ln2b, out, nullptr);
}
